// GraphMatchingLayer_56573309223899
// MI455X (gfx1250) — matrix operand order repaired and checked by reading, not compiled, not run on hardware
//
#include <hip/hip_runtime.h>
#include <hip/hip_bf16.h>

typedef __attribute__((ext_vector_type(16))) _Float16 v16h;
typedef __attribute__((ext_vector_type(8)))  _Float16 v8h;
typedef __attribute__((ext_vector_type(8)))  float    v8f;

#define HID   64
#define EDIM  32
#define WAVES 8
#define FRAG_HALVES 512   // 32 lanes * 16 halves per WMMA B fragment

// Fragment table (all weight matrices are K x 64, 4 n-tiles each):
//   ew1: K=160 -> 5 kt * 4 nt = 20 frags   [ 0..19]
//   ew2: K= 64 -> 2 kt * 4 nt =  8 frags   [20..27]
//   nw1: K=128 -> 4 kt * 4 nt = 16 frags   [28..43]
//   nw2: K= 64 -> 2 kt * 4 nt =  8 frags   [44..51]
#define EW1_F 0
#define EW2_F 20
#define NW1_F 28
#define NW2_F 44
#define NFRAG 52

// ---- A-fragment loaders -----------------------------------------------------
// 16-bit A-matrix 16x32 layout (ISA 7.12.2): lane l (idx=l&15, hi=l>>4) holds
// row M=idx, K = hi*8 + {0..7} and hi*8 + 16 + {0..7}  -> two contiguous 16B runs.
__device__ __forceinline__ v16h frag_a_from_h16(const _Float16* p, int hi) {
  v8h lo = *(const v8h*)(p + hi * 8);
  v8h hv = *(const v8h*)(p + hi * 8 + 16);
  v16h a;
#pragma unroll
  for (int j = 0; j < 8; ++j) { a[j] = lo[j]; a[j + 8] = hv[j]; }
  return a;
}

__device__ __forceinline__ v16h frag_a_from_f32(const float* p, int hi) {
  v8f lo = *(const v8f*)(p + hi * 8);
  v8f hv = *(const v8f*)(p + hi * 8 + 16);
  v16h a;
#pragma unroll
  for (int j = 0; j < 8; ++j) {
    a[j]     = (_Float16)lo[j];
    a[j + 8] = (_Float16)hv[j];
  }
  return a;
}

// ---- prep: x -> f16, zero aggregation buffer (lives in d_out) ---------------
__global__ void prep_kernel(const float* __restrict__ x, _Float16* __restrict__ xh,
                            float* __restrict__ agg, int total) {
  int i = blockIdx.x * blockDim.x + threadIdx.x;
  int stride = gridDim.x * blockDim.x;
  for (; i < total; i += stride) {
    xh[i]  = (_Float16)x[i];
    agg[i] = 0.0f;
  }
}

// ---- pack: f32 weights -> f16 WMMA B-fragment layout ------------------------
// 16-bit B-matrix 32x16: lane (idx,hi) holds column N=idx, K = hi*8 + {0..7} and hi*8 + 16 + {0..7}.
__global__ void pack_kernel(const float* __restrict__ ew1, const float* __restrict__ ew2,
                            const float* __restrict__ nw1, const float* __restrict__ nw2,
                            _Float16* __restrict__ wp) {
  int g = blockIdx.x * blockDim.x + threadIdx.x;
  if (g >= NFRAG * 32) return;
  int frag = g >> 5, lane = g & 31;
  int idx = lane & 15, hi = lane >> 4;
  const float* w; int fi;
  if (frag < EW2_F)      { w = ew1; fi = frag; }
  else if (frag < NW1_F) { w = ew2; fi = frag - EW2_F; }
  else if (frag < NW2_F) { w = nw1; fi = frag - NW1_F; }
  else                   { w = nw2; fi = frag - NW2_F; }
  int kt = fi >> 2, nt = fi & 3;
  int col = nt * 16 + idx;
  v16h b;
#pragma unroll
  for (int j = 0; j < 16; ++j)
    b[j] = (_Float16)w[(size_t)(kt * 32 + hi * 8 + (j & 7) + (j >> 3) * 16) * HID + col];
  *(v16h*)(wp + (size_t)frag * FRAG_HALVES + lane * 16) = b;
}

// ---- edge kernel: one wave per 16-edge tile ---------------------------------
__global__ __launch_bounds__(256) void edge_kernel(
    const _Float16* __restrict__ xh, const int* __restrict__ ei,
    const float* __restrict__ ea, const _Float16* __restrict__ wp,
    const float* __restrict__ eb1, const float* __restrict__ eb2,
    float* __restrict__ agg, int E) {
  __shared__ __align__(16) _Float16 hsm[WAVES][16 * HID];
  const int wave = threadIdx.x >> 5;
  const int lane = threadIdx.x & 31;
  const int idx  = lane & 15;
  const int hi   = lane >> 4;
  const int base = (blockIdx.x * WAVES + wave) * 16;
  if (base >= E) return;                    // wave-uniform: EXEC stays all-1s

  const int e  = base + idx;
  const int ec = (e < E) ? e : 0;
  const int r  = ei[ec];                    // source node of this lane's edge-row
  const int c  = ei[E + ec];                // dest node

  int rowi[8];                              // scatter targets for the C rows we hold
#pragma unroll
  for (int i = 0; i < 8; ++i) {
    int em = base + hi * 8 + i;
    rowi[i] = ei[(em < E) ? em : 0];
  }

  // prefetch next tile's edge_attr rows
  if (e + 16 < E) __builtin_prefetch(ea + (size_t)(e + 16) * EDIM, 0, 0);

  // GEMM1: [16,160] @ ew1 -> [16,64], bias preloaded (C column fixed per lane)
  v8f acc[4];
#pragma unroll
  for (int nt = 0; nt < 4; ++nt) {
    float b = eb1[nt * 16 + idx];
#pragma unroll
    for (int i = 0; i < 8; ++i) acc[nt][i] = b;
  }

  const _Float16* w1 = wp + (size_t)EW1_F * FRAG_HALVES;
#pragma unroll
  for (int kt = 0; kt < 5; ++kt) {
    v16h a;
    if (kt < 2)      a = frag_a_from_h16(xh + (size_t)r * HID + kt * 32, hi);
    else if (kt < 4) a = frag_a_from_h16(xh + (size_t)c * HID + (kt - 2) * 32, hi);
    else             a = frag_a_from_f32(ea + (size_t)ec * EDIM, hi);
    if (e >= E) {                           // zero contributions of tail rows
#pragma unroll
      for (int j = 0; j < 16; ++j) a[j] = (_Float16)0.0f;
    }
#pragma unroll
    for (int nt = 0; nt < 4; ++nt) {
      v16h b = *(const v16h*)(w1 + (size_t)(kt * 4 + nt) * FRAG_HALVES + lane * 16);
      acc[nt] = __builtin_amdgcn_wmma_f32_16x16x32_f16(
          false, a, false, b, (short)0, acc[nt], false, false);
    }
  }

  // relu + transpose C-layout (column-per-lane) -> A-layout via LDS
  _Float16* hs = hsm[wave];
#pragma unroll
  for (int nt = 0; nt < 4; ++nt)
#pragma unroll
    for (int i = 0; i < 8; ++i) {
      float v = acc[nt][i];
      v = v > 0.0f ? v : 0.0f;
      hs[(hi * 8 + i) * HID + nt * 16 + idx] = (_Float16)v;
    }
  asm volatile("s_wait_dscnt 0" ::: "memory");   // in-wave LDS RAW

  // GEMM2: [16,64] @ ew2 -> messages [16,64]
  v8f acc2[4];
#pragma unroll
  for (int nt = 0; nt < 4; ++nt) {
    float b = eb2[nt * 16 + idx];
#pragma unroll
    for (int i = 0; i < 8; ++i) acc2[nt][i] = b;
  }
  const _Float16* w2 = wp + (size_t)EW2_F * FRAG_HALVES;
#pragma unroll
  for (int kt = 0; kt < 2; ++kt) {
    v16h a = frag_a_from_h16(hs + idx * HID + kt * 32, hi);
#pragma unroll
    for (int nt = 0; nt < 4; ++nt) {
      v16h b = *(const v16h*)(w2 + (size_t)(kt * 4 + nt) * FRAG_HALVES + lane * 16);
      acc2[nt] = __builtin_amdgcn_wmma_f32_16x16x32_f16(
          false, a, false, b, (short)0, acc2[nt], false, false);
    }
  }

  // scatter-add messages into agg[row]
#pragma unroll
  for (int nt = 0; nt < 4; ++nt)
#pragma unroll
    for (int i = 0; i < 8; ++i) {
      int em = base + hi * 8 + i;
      if (em < E)
        atomicAdd(agg + (size_t)rowi[i] * HID + nt * 16 + idx, acc2[nt][i]);
    }
}

// ---- node kernel: one wave per 16-node tile ---------------------------------
__global__ __launch_bounds__(256) void node_kernel(
    const _Float16* __restrict__ xh, float* __restrict__ io,
    const _Float16* __restrict__ wp, const float* __restrict__ nb1,
    const float* __restrict__ nb2, int N) {
  __shared__ __align__(16) _Float16 hsm[WAVES][16 * HID];
  const int wave = threadIdx.x >> 5;
  const int lane = threadIdx.x & 31;
  const int idx  = lane & 15;
  const int hi   = lane >> 4;
  const int base = (blockIdx.x * WAVES + wave) * 16;
  if (base >= N) return;

  int rrow = base + idx;
  if (rrow >= N) rrow = base;               // clamp inside own tile (no cross-tile race)

  v8f acc[4];
#pragma unroll
  for (int nt = 0; nt < 4; ++nt) {
    float b = nb1[nt * 16 + idx];
#pragma unroll
    for (int i = 0; i < 8; ++i) acc[nt][i] = b;
  }

  // A = [x (f16) | agg (f32, read from io before we overwrite)] , K = 128
  const _Float16* w1 = wp + (size_t)NW1_F * FRAG_HALVES;
#pragma unroll
  for (int kt = 0; kt < 4; ++kt) {
    v16h a;
    if (kt < 2) a = frag_a_from_h16(xh + (size_t)rrow * HID + kt * 32, hi);
    else        a = frag_a_from_f32(io + (size_t)rrow * HID + (kt - 2) * 32, hi);
#pragma unroll
    for (int nt = 0; nt < 4; ++nt) {
      v16h b = *(const v16h*)(w1 + (size_t)(kt * 4 + nt) * FRAG_HALVES + lane * 16);
      acc[nt] = __builtin_amdgcn_wmma_f32_16x16x32_f16(
          false, a, false, b, (short)0, acc[nt], false, false);
    }
  }

  _Float16* hs = hsm[wave];
#pragma unroll
  for (int nt = 0; nt < 4; ++nt)
#pragma unroll
    for (int i = 0; i < 8; ++i) {
      float v = acc[nt][i];
      v = v > 0.0f ? v : 0.0f;
      hs[(hi * 8 + i) * HID + nt * 16 + idx] = (_Float16)v;
    }
  asm volatile("s_wait_dscnt 0" ::: "memory");

  v8f acc2[4];
#pragma unroll
  for (int nt = 0; nt < 4; ++nt) {
    float b = nb2[nt * 16 + idx];
#pragma unroll
    for (int i = 0; i < 8; ++i) acc2[nt][i] = b;
  }
  const _Float16* w2 = wp + (size_t)NW2_F * FRAG_HALVES;
#pragma unroll
  for (int kt = 0; kt < 2; ++kt) {
    v16h a = frag_a_from_h16(hs + idx * HID + kt * 32, hi);
#pragma unroll
    for (int nt = 0; nt < 4; ++nt) {
      v16h b = *(const v16h*)(w2 + (size_t)(kt * 4 + nt) * FRAG_HALVES + lane * 16);
      acc2[nt] = __builtin_amdgcn_wmma_f32_16x16x32_f16(
          false, a, false, b, (short)0, acc2[nt], false, false);
    }
  }

  // store final output (overwrites agg rows of this tile — reads already done)
#pragma unroll
  for (int nt = 0; nt < 4; ++nt)
#pragma unroll
    for (int i = 0; i < 8; ++i) {
      int m = base + hi * 8 + i;
      if (m < N)
        io[(size_t)m * HID + nt * 16 + idx] = acc2[nt][i];
    }
}

extern "C" void kernel_launch(void* const* d_in, const int* in_sizes, int n_in,
                              void* d_out, int out_size, void* d_ws, size_t ws_size,
                              hipStream_t stream) {
  const float* x   = (const float*)d_in[0];
  const int*   ei  = (const int*)d_in[1];
  const float* ea  = (const float*)d_in[2];
  const float* ew1 = (const float*)d_in[3];
  const float* eb1 = (const float*)d_in[4];
  const float* ew2 = (const float*)d_in[5];
  const float* eb2 = (const float*)d_in[6];
  const float* nw1 = (const float*)d_in[7];
  const float* nb1 = (const float*)d_in[8];
  const float* nw2 = (const float*)d_in[9];
  const float* nb2 = (const float*)d_in[10];
  float* io = (float*)d_out;   // agg accumulator, then final output

  const int N = in_sizes[0] / HID;
  const int E = in_sizes[2] / EDIM;

  // ws layout: [ xh : N*HID f16 ][ packed weights : 52 KB ]
  _Float16* xh = (_Float16*)d_ws;
  size_t xh_bytes = ((size_t)N * HID * sizeof(_Float16) + 255) & ~(size_t)255;
  _Float16* wp = (_Float16*)((char*)d_ws + xh_bytes);

  const int total = N * HID;
  int prep_blocks = (total + 256 * 8 - 1) / (256 * 8);
  prep_kernel<<<prep_blocks, 256, 0, stream>>>(x, xh, io, total);

  pack_kernel<<<(NFRAG * 32 + 255) / 256, 256, 0, stream>>>(ew1, ew2, nw1, nw2, wp);

  int etiles = (E + 15) / 16;
  edge_kernel<<<(etiles + WAVES - 1) / WAVES, 32 * WAVES, 0, stream>>>(
      xh, ei, ea, wp, eb1, eb2, io, E);

  int ntiles = (N + 15) / 16;
  node_kernel<<<(ntiles + WAVES - 1) / WAVES, 32 * WAVES, 0, stream>>>(
      xh, io, wp, nb1, nb2, N);
}
